// A1SparseCoding_74801150427781
// MI455X (gfx1250) — compile-verified, matrix operand order checked by reading, not run on hardware
//
#include <hip/hip_runtime.h>
#include <math.h>

typedef __bf16 bf16;
typedef bf16  v16bf __attribute__((ext_vector_type(16)));
typedef bf16  v8bf  __attribute__((ext_vector_type(8)));
typedef float v8f   __attribute__((ext_vector_type(8)));
typedef float v2f   __attribute__((ext_vector_type(2)));

#define N_BASES   128
#define SEG       3200
#define NFFT      512
#define HOP       160
#define NFREQ     257
#define NFRAMES   17
#define SPEC_DIM  4369
#define BATCH     8192
#define KPAD      4384     /* 137 * 32, zero-padded K for bf16 WMMA */
#define KSTEPS    (KPAD / 32)
#define NCOL      514      /* 257 freqs * (re, im) interleaved      */
#define NCOLP     528      /* 33 * 16 N-tiles                        */
#define LAM       0.5f
#define INV_TAU   0.1f
#define ITERS     50
#define TWO_PI    6.28318530717958647692f

__device__ __forceinline__ float softshrink_f(float u) {
    return copysignf(fmaxf(fabsf(u) - LAM, 0.0f), u);
}

// 16x16x32 bf16 fragment: element e covers k = base + (e&8)*2 + hi*8 + (e&7)
__device__ __forceinline__ v16bf pack16(v8bf lo, v8bf hi8) {
    return __builtin_shufflevector(lo, hi8, 0,1,2,3,4,5,6,7,8,9,10,11,12,13,14,15);
}

__device__ __forceinline__ v16bf ld_frag(const bf16* row, int ks, int hi) {
    const int off = ks * 32 + hi * 8;
    return pack16(*(const v8bf*)(row + off), *(const v8bf*)(row + off + 16));
}

__device__ __forceinline__ v16bf ld_frag_nt(const bf16* row, int ks, int hi) {
    const int off = ks * 32 + hi * 8;
    return pack16(__builtin_nontemporal_load((const v8bf*)(row + off)),
                  __builtin_nontemporal_load((const v8bf*)(row + off + 16)));
}

// ---------------------------------------------------------------------------
// Kernel 1: windowed-DFT basis, transposed:  basisT[n][k] = w(k)*trig(f(n),k)
// n even -> cos (real part), n odd -> -sin (imag part). Window folded in.
// ---------------------------------------------------------------------------
__global__ void k_basis(bf16* __restrict__ basisT) {
    int i = blockIdx.x * blockDim.x + threadIdx.x;
    if (i >= NCOLP * NFFT) return;
    int n = i / NFFT, k = i - n * NFFT;
    float v = 0.0f;
    if (n < NCOL) {
        int f  = n >> 1;
        int ph = (f * k) & (NFFT - 1);                 // exact phase fold
        float ang = TWO_PI * (float)ph / (float)NFFT;
        float w   = 0.5f * (1.0f - cosf(TWO_PI * (float)k / (float)NFFT));
        v = w * ((n & 1) ? -sinf(ang) : cosf(ang));
    }
    basisT[i] = (bf16)v;
}

// ---------------------------------------------------------------------------
// Kernel 2: dictionary -> bf16, rows padded to KPAD with zeros
// ---------------------------------------------------------------------------
__global__ void k_prep_D(const float* __restrict__ D, bf16* __restrict__ Dbf) {
    int i = blockIdx.x * blockDim.x + threadIdx.x;
    if (i >= N_BASES * KPAD) return;
    int n = i / KPAD, k = i - n * KPAD;
    float v = (k < SPEC_DIM) ? D[n * SPEC_DIM + k] : 0.0f;
    Dbf[i] = (bf16)v;
}

// ---------------------------------------------------------------------------
// Kernel 3: gram = D @ D^T - I   (128x128, fp32, symmetric). 143 MFLOP: trivial.
// ---------------------------------------------------------------------------
__global__ void k_gram(const float* __restrict__ D, float* __restrict__ gram) {
    int i = blockIdx.x * blockDim.x + threadIdx.x;   // 16384 threads
    int r = i >> 7, c = i & 127;
    const float* a = D + (size_t)r * SPEC_DIM;
    const float* b = D + (size_t)c * SPEC_DIM;
    float s = 0.0f;
    for (int k = 0; k < SPEC_DIM; ++k) s = fmaf(a[k], b[k], s);
    if (r == c) s -= 1.0f;
    gram[i] = s;
}

// ---------------------------------------------------------------------------
// Kernel 4: STFT magnitude (DFT-as-GEMM via bf16 WMMA) + per-sample normalize.
// One sample per block, 2 waves: wave0 = frames 0..15, wave1 = frame 16 (+pad).
// Audio staged global->LDS via CDNA5 async-to-LDS DMA (ASYNCcnt).
// B-panel streamed with a 4-slot rotating buffer; sched_barrier(0) pins the
// load-ahead shape so the scheduler cannot re-serialize it.
// ---------------------------------------------------------------------------
__global__ __launch_bounds__(64) void
k_stft(const float* __restrict__ audio, const bf16* __restrict__ basisT,
       bf16* __restrict__ specn) {
    __shared__ __align__(16) float au[SEG];
    __shared__ float spec[SPEC_DIM];
    __shared__ float red0[64], red1[64];
    __shared__ float stats[2];

    const int b    = blockIdx.x;
    const int tid  = threadIdx.x;
    const int wave = tid >> 5, lane = tid & 31;
    const int hi   = (lane >> 4) & 1;
    const int mrow = lane & 15;

    // ---- async DMA: audio row -> LDS (16B chunks, no VGPR round-trip) ----
    for (int i = tid * 4; i < SEG; i += 64 * 4) {
        unsigned           lds_off = (unsigned)(uintptr_t)(&au[i]);
        unsigned long long gaddr   = (unsigned long long)(audio + (size_t)b * SEG + i);
        asm volatile("global_load_async_to_lds_b128 %0, %1, off"
                     :: "v"(lds_off), "v"(gaddr) : "memory");
    }
    asm volatile("s_wait_asynccnt 0x0" ::: "memory");
    __syncthreads();

    // Preload all A fragments (raw audio frames; window lives in basis).
    const int  t    = wave * 16 + mrow;
    const bool tv   = (t < NFRAMES);
    const int  tb   = tv ? t * HOP : 0;
    v16bf afrag[16];
#pragma unroll
    for (int ks = 0; ks < 16; ++ks) {
        const int kb = ks * 32 + hi * 8;
#pragma unroll
        for (int e = 0; e < 16; ++e) {
            int k = kb + (e & 8) * 2 + (e & 7);
            float x = tv ? au[tb + k] : 0.0f;
            afrag[ks][e] = (bf16)x;
        }
    }

    for (int nt = 0; nt < NCOLP / 16; ++nt) {
        const int n = nt * 16 + mrow;
        const bf16* brow = basisT + (size_t)n * NFFT;
        v16bf B0 = ld_frag(brow, 0, hi);
        v16bf B1 = ld_frag(brow, 1, hi);
        v16bf B2 = ld_frag(brow, 2, hi);
        v16bf B3 = ld_frag(brow, 3, hi);
        __builtin_amdgcn_sched_barrier(0);
        v8f c = {};
#pragma unroll
        for (int ks = 0; ks < 16; ks += 4) {
            c = __builtin_amdgcn_wmma_f32_16x16x32_bf16(
                    false, afrag[ks + 0], false, B0, (short)0, c, false, false);
            if (ks + 4 < 16) B0 = ld_frag(brow, ks + 4, hi);
            __builtin_amdgcn_sched_barrier(0);
            c = __builtin_amdgcn_wmma_f32_16x16x32_bf16(
                    false, afrag[ks + 1], false, B1, (short)0, c, false, false);
            if (ks + 5 < 16) B1 = ld_frag(brow, ks + 5, hi);
            __builtin_amdgcn_sched_barrier(0);
            c = __builtin_amdgcn_wmma_f32_16x16x32_bf16(
                    false, afrag[ks + 2], false, B2, (short)0, c, false, false);
            if (ks + 6 < 16) B2 = ld_frag(brow, ks + 6, hi);
            __builtin_amdgcn_sched_barrier(0);
            c = __builtin_amdgcn_wmma_f32_16x16x32_bf16(
                    false, afrag[ks + 3], false, B3, (short)0, c, false, false);
            if (ks + 7 < 16) B3 = ld_frag(brow, ks + 7, hi);
            __builtin_amdgcn_sched_barrier(0);
        }
        // combine (re, im) neighbor lanes -> magnitude, scatter to spec[f*17+t]
#pragma unroll
        for (int r = 0; r < 8; ++r) {
            float v  = c[r];
            float ov = __shfl_xor(v, 1);
            int   m  = r + hi * 8;
            int   tt = wave * 16 + m;
            if (tt < NFRAMES && n < NCOL && !(n & 1)) {
                spec[(n >> 1) * NFRAMES + tt] = sqrtf(v * v + ov * ov);
            }
        }
    }
    __syncthreads();

    float s = 0.0f, s2 = 0.0f;
    for (int i = tid; i < SPEC_DIM; i += 64) { float v = spec[i]; s += v; s2 += v * v; }
    red0[tid] = s; red1[tid] = s2;
    __syncthreads();
    if (tid == 0) {
        float a0 = 0.0f, a1 = 0.0f;
        for (int i = 0; i < 64; ++i) { a0 += red0[i]; a1 += red1[i]; }
        float mean = a0 / (float)SPEC_DIM;
        float var  = fmaxf((a1 - (float)SPEC_DIM * mean * mean) / (float)(SPEC_DIM - 1), 0.0f);
        stats[0] = mean;
        stats[1] = 1.0f / (sqrtf(var) + 1e-8f);
    }
    __syncthreads();
    float mean = stats[0], inv = stats[1];
    for (int i = tid; i < KPAD; i += 64) {
        float v = (i < SPEC_DIM) ? (spec[i] - mean) * inv : 0.0f;
        specn[(size_t)b * KPAD + i] = (bf16)v;
    }
}

// ---------------------------------------------------------------------------
// Kernel 5: b = specn @ D^T  (M=8192, N=128, K=4384 padded) via bf16 WMMA.
// One wave per 16x16 tile; 4-slot rotating A/B buffers (137 = 4*34 + 1 steps)
// pinned with sched_barrier; specn is read exactly once -> nontemporal.
// ---------------------------------------------------------------------------
__global__ __launch_bounds__(32) void
k_proj(const bf16* __restrict__ specn, const bf16* __restrict__ Dbf,
       float* __restrict__ bvec) {
    const int mt = blockIdx.x, nt = blockIdx.y;
    const int lane = threadIdx.x;
    const int hi   = (lane >> 4) & 1;
    const int mrow = lane & 15;
    const bf16* arow = specn + (size_t)(mt * 16 + mrow) * KPAD;
    const bf16* brow = Dbf   + (size_t)(nt * 16 + mrow) * KPAD;

    v16bf A0 = ld_frag_nt(arow, 0, hi), B0 = ld_frag(brow, 0, hi);
    v16bf A1 = ld_frag_nt(arow, 1, hi), B1 = ld_frag(brow, 1, hi);
    v16bf A2 = ld_frag_nt(arow, 2, hi), B2 = ld_frag(brow, 2, hi);
    v16bf A3 = ld_frag_nt(arow, 3, hi), B3 = ld_frag(brow, 3, hi);
    __builtin_amdgcn_sched_barrier(0);
    v8f c = {};
#pragma unroll 1
    for (int ks = 0; ks + 4 <= KSTEPS; ks += 4) {
        int n0 = (ks + 4 < KSTEPS) ? ks + 4 : KSTEPS - 1;
        int n1 = (ks + 5 < KSTEPS) ? ks + 5 : KSTEPS - 1;
        int n2 = (ks + 6 < KSTEPS) ? ks + 6 : KSTEPS - 1;
        int n3 = (ks + 7 < KSTEPS) ? ks + 7 : KSTEPS - 1;
        c = __builtin_amdgcn_wmma_f32_16x16x32_bf16(
                false, A0, false, B0, (short)0, c, false, false);
        A0 = ld_frag_nt(arow, n0, hi); B0 = ld_frag(brow, n0, hi);
        __builtin_amdgcn_sched_barrier(0);
        c = __builtin_amdgcn_wmma_f32_16x16x32_bf16(
                false, A1, false, B1, (short)0, c, false, false);
        A1 = ld_frag_nt(arow, n1, hi); B1 = ld_frag(brow, n1, hi);
        __builtin_amdgcn_sched_barrier(0);
        c = __builtin_amdgcn_wmma_f32_16x16x32_bf16(
                false, A2, false, B2, (short)0, c, false, false);
        A2 = ld_frag_nt(arow, n2, hi); B2 = ld_frag(brow, n2, hi);
        __builtin_amdgcn_sched_barrier(0);
        c = __builtin_amdgcn_wmma_f32_16x16x32_bf16(
                false, A3, false, B3, (short)0, c, false, false);
        A3 = ld_frag_nt(arow, n3, hi); B3 = ld_frag(brow, n3, hi);
        __builtin_amdgcn_sched_barrier(0);
    }
    // tail: step 136 (slot 0, loaded during the last main-loop body)
    c = __builtin_amdgcn_wmma_f32_16x16x32_bf16(
            false, A0, false, B0, (short)0, c, false, false);

#pragma unroll
    for (int r = 0; r < 8; ++r) {
        bvec[(size_t)(mt * 16 + r + hi * 8) * N_BASES + nt * 16 + mrow] = c[r];
    }
}

// ---------------------------------------------------------------------------
// Kernel 6: LCA fixed-point, 50 iters, fp32 WMMA 16x16x4.
// One wave per 16-row tile; u/a state in LDS; gram (symmetric) streamed from
// L2/WGP$ as contiguous v2f B-fragments (gram[k][n] == gram[n][k]).
// ---------------------------------------------------------------------------
__global__ __launch_bounds__(32) void
k_lca(const float* __restrict__ gram, const float* __restrict__ bvec,
      float* __restrict__ out) {
    __shared__ float us[16 * N_BASES];
    __shared__ float as[16 * N_BASES];
    const int mt   = blockIdx.x;
    const int lane = threadIdx.x;
    const int hi   = (lane >> 4) & 1;
    const int mrow = lane & 15;

    // b in C/D-fragment layout, kept in registers across all iterations
    v8f bfrag[8];
#pragma unroll
    for (int nt = 0; nt < 8; ++nt)
#pragma unroll
        for (int r = 0; r < 8; ++r)
            bfrag[nt][r] =
                bvec[(size_t)(mt * 16 + r + hi * 8) * N_BASES + nt * 16 + mrow];

    for (int i = lane; i < 16 * N_BASES; i += 32) us[i] = 0.0f;
    __syncthreads();

    for (int it = 0; it < ITERS; ++it) {
        for (int i = lane; i < 16 * N_BASES; i += 32) as[i] = softshrink_f(us[i]);
        __syncthreads();
#pragma unroll 1
        for (int nt = 0; nt < 8; ++nt) {
            v8f c = {};
#pragma unroll
            for (int k4 = 0; k4 < 32; ++k4) {
                const int kb = k4 * 4 + hi * 2;
                v2f av = *(const v2f*)(&as[mrow * N_BASES + kb]);
                v2f bv = *(const v2f*)(gram + (size_t)(nt * 16 + mrow) * N_BASES + kb);
                c = __builtin_amdgcn_wmma_f32_16x16x4_f32(
                        false, av, false, bv, (short)0, c, false, false);
            }
#pragma unroll
            for (int r = 0; r < 8; ++r) {
                const int idx = (r + hi * 8) * N_BASES + nt * 16 + mrow;
                float uu = us[idx];
                us[idx] = uu + (bfrag[nt][r] - uu - c[r]) * INV_TAU;
            }
        }
        __syncthreads();
    }

#pragma unroll
    for (int nt = 0; nt < 8; ++nt)
#pragma unroll
        for (int r = 0; r < 8; ++r) {
            const int idx = (r + hi * 8) * N_BASES + nt * 16 + mrow;
            out[(size_t)(mt * 16 + r + hi * 8) * N_BASES + nt * 16 + mrow] =
                softshrink_f(us[idx]);
        }
}

// ---------------------------------------------------------------------------
extern "C" void kernel_launch(void* const* d_in, const int* in_sizes, int n_in,
                              void* d_out, int out_size, void* d_ws, size_t ws_size,
                              hipStream_t stream) {
    (void)in_sizes; (void)n_in; (void)out_size; (void)ws_size;
    const float* audio = (const float*)d_in[0];   // (8192, 3200) f32
    const float* D     = (const float*)d_in[1];   // (128, 4369)  f32
    float*       out   = (float*)d_out;           // (8192, 128)  f32

    char* ws = (char*)d_ws;
    size_t off = 0;
    auto carve = [&](size_t bytes) -> char* {
        char* p = ws + off;
        off = (off + bytes + 255) & ~(size_t)255;
        return p;
    };
    bf16*  basisT = (bf16*) carve((size_t)NCOLP * NFFT * sizeof(bf16));       // 540 KB
    bf16*  Dbf    = (bf16*) carve((size_t)N_BASES * KPAD * sizeof(bf16));     // 1.1 MB
    float* gram   = (float*)carve((size_t)N_BASES * N_BASES * sizeof(float)); // 64 KB
    bf16*  specn  = (bf16*) carve((size_t)BATCH * KPAD * sizeof(bf16));       // 71.8 MB
    float* bvec   = (float*)carve((size_t)BATCH * N_BASES * sizeof(float));   // 4 MB

    k_basis <<<(NCOLP * NFFT + 255) / 256, 256, 0, stream>>>(basisT);
    k_prep_D<<<(N_BASES * KPAD + 255) / 256, 256, 0, stream>>>(D, Dbf);
    k_gram  <<<(N_BASES * N_BASES) / 256, 256, 0, stream>>>(D, gram);
    k_stft  <<<BATCH, 64, 0, stream>>>(audio, basisT, specn);
    k_proj  <<<dim3(BATCH / 16, N_BASES / 16), 32, 0, stream>>>(specn, Dbf, bvec);
    k_lca   <<<BATCH / 16, 32, 0, stream>>>(gram, bvec, out);
}
